// Model_15710990369331
// MI455X (gfx1250) — matrix operand order repaired and checked by reading, not compiled, not run on hardware
//
#include <hip/hip_runtime.h>

#define N_ALL 60000
#define N_G2  40000
#define E1_N  600000
#define E2_N  320000
#define R_N   8
#define NB_N  4
#define D_N   200
#define L_N   1024
#define S_N   256

#define KPAD  224            // 7 * 32 (K padded with zeros)
#define NPAD  208            // 13 * 16 (N padded with zeros)
#define KCH   7
#define MT    (N_G2 / 16)    // 2500 M-tiles (exact)
#define NT    (NPAD / 16)    // 13 N-tiles
#define NTH   7              // N-tiles handled per gridDim.y half (7 then 6)

typedef __attribute__((ext_vector_type(16))) __bf16 v16bf;
typedef __attribute__((ext_vector_type(8)))  float  v8f;

__device__ __forceinline__ unsigned short f2bf(float f) {
  union { float f; unsigned int u; } v; v.f = f;
  unsigned int u = v.u;
  u += 0x7FFFu + ((u >> 16) & 1u);   // round-to-nearest-even
  return (unsigned short)(u >> 16);
}

// ---------------- concept layer (graph 1) ----------------
__global__ void concept_edge_kernel(const int* __restrict__ ei,
                                    const float* __restrict__ x,
                                    float* __restrict__ agg,
                                    float* __restrict__ deg) {
  int e = blockIdx.x;
  int src = ei[e];
  int dst = ei[E1_N + e];
  const float* xr = x + (size_t)src * D_N;
  float* ar = agg + (size_t)dst * D_N;
  for (int f = threadIdx.x; f < D_N; f += blockDim.x)
    atomicAdd(ar + f, xr[f]);
  if (threadIdx.x == 0) atomicAdd(deg + dst, 1.0f);
}

__global__ void concept_norm_kernel(const float* __restrict__ x,
                                    const float* __restrict__ agg,
                                    const float* __restrict__ deg,
                                    float* __restrict__ xg1) {
  int i = blockIdx.x * blockDim.x + threadIdx.x;
  if (i >= N_ALL * D_N) return;
  int n = i / D_N;
  float v = (x[i] + agg[i]) / (1.0f + deg[n]);
  xg1[i] = fmaxf(v, 0.0f);
}

// ---------------- RGCN helpers ----------------
__global__ void cnt_kernel(const int* __restrict__ ei,
                           const int* __restrict__ et,
                           float* __restrict__ cnt) {
  int e = blockIdx.x * blockDim.x + threadIdx.x;
  if (e >= E2_N) return;
  int dst = ei[E2_N + e];
  atomicAdd(cnt + dst * R_N + et[e], 1.0f);
}

// Build W in per-lane WMMA-B swizzled layout, bf16, zero-padded.
// Bsw[rmat][kc][nt][lane][j] holds W[rmat][k=kc*32+(lane>>4)*16+j][h=nt*16+(lane&15)]
// so each lane's 16 B-halves for one K-chunk are CONTIGUOUS (-> two b128 loads).
__global__ void build_w_kernel(const float* __restrict__ basis,
                               const float* __restrict__ comb,
                               const float* __restrict__ root,
                               unsigned short* __restrict__ Wb) {
  int i = blockIdx.x * blockDim.x + threadIdx.x;
  const int PER_R = KCH * NT * 512;           // halves per matrix
  if (i >= (R_N + 1) * PER_R) return;
  int rmat = i / PER_R;
  int rem  = i - rmat * PER_R;
  int kc   = rem / (NT * 512);
  int rem2 = rem - kc * (NT * 512);
  int nt   = rem2 / 512;
  int t    = rem2 - nt * 512;
  int lane = t >> 4;
  int j    = t & 15;
  int k = kc * 32 + (lane >> 4) * 16 + j;
  int h = nt * 16 + (lane & 15);
  float v = 0.0f;
  if (k < D_N && h < D_N) {
    if (rmat < R_N) {
      float s = 0.0f;
#pragma unroll
      for (int b = 0; b < NB_N; ++b)
        s += comb[rmat * NB_N + b] * basis[((size_t)b * D_N + k) * D_N + h];
      v = s;
    } else {
      v = root[k * D_N + h];
    }
  }
  Wb[i] = f2bf(v);
}

__global__ void to_bf16_kernel(const float* __restrict__ x,
                               unsigned short* __restrict__ A) {
  int i = blockIdx.x * blockDim.x + threadIdx.x;
  if (i >= N_G2 * KPAD) return;
  int n = i / KPAD;
  int k = i - n * KPAD;
  A[i] = (k < D_N) ? f2bf(x[(size_t)n * D_N + k]) : (unsigned short)0;
}

// ---------------- bf16 WMMA GEMM: Y[40000, NPAD] = A[40000, KPAD] @ W ----------------
// Block = 8 waves = 8 M-tiles. gridDim.y = 2 splits the 13 N-tiles (7 + 6).
// Swizzled W slice staged in LDS once per block; A tile held in registers for all N-tiles.
__global__ __launch_bounds__(256)
void gemm_bf16_kernel(const unsigned short* __restrict__ A,
                      const unsigned short* __restrict__ Bsw,  // [KCH][NT][512] halves
                      float* __restrict__ Y) {
  __shared__ __align__(16) unsigned short lB[KCH * NTH * 512]; // 50,176 B

  int ntbase = blockIdx.y * NTH;
  int ntcnt  = (blockIdx.y == 0) ? NTH : (NT - NTH);           // 7 or 6

  // cooperative stage of this block's W slice into LDS (b128 granularity)
  {
    const uint4* g = (const uint4*)Bsw;
    uint4* l = (uint4*)lB;
    int total = KCH * ntcnt * 64;                              // uint4 count
    for (int i = threadIdx.x; i < total; i += 256) {
      int kc = i / (ntcnt * 64);
      int w  = i - kc * (ntcnt * 64);
      l[kc * (NTH * 64) + w] = g[kc * (NT * 64) + ntbase * 64 + w];
    }
  }
  __syncthreads();

  int wave = threadIdx.x >> 5;
  int lane = threadIdx.x & 31;
  int mt = blockIdx.x * 8 + wave;                              // wave-uniform
  if (mt >= MT) return;
  int r  = lane & 15;
  int hi = lane >> 4;

  // A 16x32 bf16, contiguous k order used here (same as Bsw): lanes 0-15 hold K {0..15}, lanes 16-31 hold K {16..31}
  const unsigned short* Arow = A + (size_t)(mt * 16 + r) * KPAD;
  union AV { v16bf v; uint4 q[2]; } a[KCH];
#pragma unroll
  for (int kc = 0; kc < KCH; ++kc) {
    a[kc].q[0] = *(const uint4*)(Arow + kc * 32 + hi * 16);
    a[kc].q[1] = *(const uint4*)(Arow + kc * 32 + hi * 16 + 8);
  }

  union BV { v16bf v; uint4 q[2]; } b;
#pragma unroll 1
  for (int ntl = 0; ntl < ntcnt; ++ntl) {
    int nt = ntbase + ntl;
    v8f acc = {};
#pragma unroll
    for (int kc = 0; kc < KCH; ++kc) {
      const uint4* lb = (const uint4*)(lB + ((kc * NTH + ntl) * 512 + lane * 16));
      b.q[0] = lb[0];
      b.q[1] = lb[1];
      acc = __builtin_amdgcn_wmma_f32_16x16x32_bf16(false, a[kc].v, false, b.v,
                                                    (short)0, acc, false, false);
    }
    // C/D layout: VGPR v holds row m = hi*8 + v, col n = r
    float* Yb = Y + (size_t)(mt * 16 + hi * 8) * NPAD + nt * 16 + r;
#pragma unroll
    for (int v = 0; v < 8; ++v)
      Yb[(size_t)v * NPAD] = acc[v];
  }
}

// scatter messages of relation r: agg[dst] += Y[src] * 1/max(cnt[dst,r],1)
__global__ void scatter_kernel(const int* __restrict__ ei,
                               const int* __restrict__ et,
                               const float* __restrict__ cnt,
                               const float* __restrict__ Y,
                               float* __restrict__ agg, int r) {
  int e = blockIdx.x;
  if (et[e] != r) return;
  int src = ei[e], dst = ei[E2_N + e];
  float norm = 1.0f / fmaxf(cnt[dst * R_N + r], 1.0f);
  const float* Ys = Y + (size_t)src * NPAD;
  float* ad = agg + (size_t)dst * D_N;
  for (int f = threadIdx.x; f < D_N; f += blockDim.x)
    atomicAdd(ad + f, Ys[f] * norm);
}

__global__ void combine_kernel(const float* __restrict__ Yroot,
                               const float* __restrict__ agg,
                               float* __restrict__ out, int do_relu) {
  int i = blockIdx.x * blockDim.x + threadIdx.x;
  if (i >= N_G2 * D_N) return;
  int n = i / D_N;
  int h = i - n * D_N;
  float v = Yroot[(size_t)n * NPAD + h] + agg[i];
  out[i] = do_relu ? fmaxf(v, 0.0f) : v;
}

// ---------------- final scores + softmax ----------------
__global__ __launch_bounds__(256)
void score_softmax_kernel(const int* __restrict__ index_list,
                          const int* __restrict__ sample_index,
                          const float* __restrict__ xg1,
                          const float* __restrict__ xg2,
                          const float* __restrict__ wts,
                          float* __restrict__ out) {
  __shared__ __align__(16) float q[D_N];
  __shared__ float red[S_N];
  int l = blockIdx.x;
  int s = threadIdx.x;
  int en = index_list[l];
  for (int d = s; d < D_N; d += S_N) {
    float e = xg2[(size_t)en * D_N + d];
    float w = fminf(fmaxf(wts[d], 0.0f), 1.0f);
    q[d] = e * e * w;
  }
  __syncthreads();
  int node = sample_index[(size_t)l * S_N + s];
  const float4* xr4 = (const float4*)(xg1 + (size_t)node * D_N);  // rows are 800 B -> 16 B aligned
  const float4* q4  = (const float4*)q;
  float acc = 0.0f;
#pragma unroll 5
  for (int d4 = 0; d4 < D_N / 4; ++d4) {
    float4 xv = xr4[d4];
    float4 qv = q4[d4];
    acc += fminf(fmaxf(xv.x, 0.0f), 1.0f) * qv.x;
    acc += fminf(fmaxf(xv.y, 0.0f), 1.0f) * qv.y;
    acc += fminf(fmaxf(xv.z, 0.0f), 1.0f) * qv.z;
    acc += fminf(fmaxf(xv.w, 0.0f), 1.0f) * qv.w;
  }
  red[s] = acc; __syncthreads();
  for (int o = S_N / 2; o > 0; o >>= 1) {
    if (s < o) red[s] = fmaxf(red[s], red[s + o]);
    __syncthreads();
  }
  float mx = red[0]; __syncthreads();
  float ex = __expf(acc - mx);
  red[s] = ex; __syncthreads();
  for (int o = S_N / 2; o > 0; o >>= 1) {
    if (s < o) red[s] += red[s + o];
    __syncthreads();
  }
  out[(size_t)l * S_N + s] = ex / red[0];
}

// ---------------- orchestration ----------------
extern "C" void kernel_launch(void* const* d_in, const int* in_sizes, int n_in,
                              void* d_out, int out_size, void* d_ws, size_t ws_size,
                              hipStream_t stream) {
  (void)in_sizes; (void)n_in; (void)out_size; (void)ws_size;

  const int*   ei2    = (const int*)d_in[0];   // (2, E2)
  const int*   et2    = (const int*)d_in[1];   // (E2,)
  const int*   ei1    = (const int*)d_in[2];   // (2, E1)
  const int*   idxl   = (const int*)d_in[3];   // (L,)
  const int*   sidx   = (const int*)d_in[4];   // (L, S)
  const float* x0     = (const float*)d_in[5]; // (N_ALL, D)
  const float* basis1 = (const float*)d_in[6];
  const float* comb1  = (const float*)d_in[7];
  const float* root1  = (const float*)d_in[8];
  const float* basis2 = (const float*)d_in[9];
  const float* comb2  = (const float*)d_in[10];
  const float* root2  = (const float*)d_in[11];
  const float* wts    = (const float*)d_in[12]; // (D,1)
  float* out = (float*)d_out;

  char* base = (char*)d_ws;
  size_t off = 0;
  auto alloc = [&](size_t bytes) -> void* {
    void* p = base + off;
    off += (bytes + 255) & ~(size_t)255;
    return p;
  };
  float*          xg1  = (float*)alloc((size_t)N_ALL * D_N * 4);
  float*          agg  = (float*)alloc((size_t)N_ALL * D_N * 4);
  float*          deg  = (float*)alloc((size_t)N_ALL * 4);
  float*          cnt  = (float*)alloc((size_t)N_G2 * R_N * 4);
  unsigned short* Ab   = (unsigned short*)alloc((size_t)N_G2 * KPAD * 2);
  unsigned short* Wb   = (unsigned short*)alloc((size_t)(R_N + 1) * KCH * NT * 512 * 2);
  float*          Y    = (float*)alloc((size_t)N_G2 * NPAD * 4);
  float*          xg2a = (float*)alloc((size_t)N_G2 * D_N * 4);
  float*          xg2b = (float*)alloc((size_t)N_G2 * D_N * 4);

  const int TPB = 256;
  const size_t WSTRIDE = (size_t)KCH * NT * 512;   // halves per matrix
  const dim3 gemm_grid((MT + 7) / 8, 2);

  // ---- concept layer over graph 1 ----
  hipMemsetAsync(agg, 0, (size_t)N_ALL * D_N * 4, stream);
  hipMemsetAsync(deg, 0, (size_t)N_ALL * 4, stream);
  concept_edge_kernel<<<E1_N, 64, 0, stream>>>(ei1, x0, agg, deg);
  concept_norm_kernel<<<(N_ALL * D_N + TPB - 1) / TPB, TPB, 0, stream>>>(x0, agg, deg, xg1);

  // ---- per-(dst,etype) counts (shared by both RGCN layers) ----
  hipMemsetAsync(cnt, 0, (size_t)N_G2 * R_N * 4, stream);
  cnt_kernel<<<(E2_N + TPB - 1) / TPB, TPB, 0, stream>>>(ei2, et2, cnt);

  // ---- RGCN layer 1 ----
  build_w_kernel<<<((R_N + 1) * (int)WSTRIDE + TPB - 1) / TPB, TPB, 0, stream>>>(basis1, comb1, root1, Wb);
  to_bf16_kernel<<<(N_G2 * KPAD + TPB - 1) / TPB, TPB, 0, stream>>>(xg1, Ab); // x_g1[:N_G2]
  hipMemsetAsync(agg, 0, (size_t)N_G2 * D_N * 4, stream);
  for (int r = 0; r < R_N; ++r) {
    gemm_bf16_kernel<<<gemm_grid, TPB, 0, stream>>>(Ab, Wb + (size_t)r * WSTRIDE, Y);
    scatter_kernel<<<E2_N, 64, 0, stream>>>(ei2, et2, cnt, Y, agg, r);
  }
  gemm_bf16_kernel<<<gemm_grid, TPB, 0, stream>>>(Ab, Wb + (size_t)R_N * WSTRIDE, Y); // x @ root1
  combine_kernel<<<(N_G2 * D_N + TPB - 1) / TPB, TPB, 0, stream>>>(Y, agg, xg2a, 1);

  // ---- RGCN layer 2 ----
  build_w_kernel<<<((R_N + 1) * (int)WSTRIDE + TPB - 1) / TPB, TPB, 0, stream>>>(basis2, comb2, root2, Wb);
  to_bf16_kernel<<<(N_G2 * KPAD + TPB - 1) / TPB, TPB, 0, stream>>>(xg2a, Ab);
  hipMemsetAsync(agg, 0, (size_t)N_G2 * D_N * 4, stream);
  for (int r = 0; r < R_N; ++r) {
    gemm_bf16_kernel<<<gemm_grid, TPB, 0, stream>>>(Ab, Wb + (size_t)r * WSTRIDE, Y);
    scatter_kernel<<<E2_N, 64, 0, stream>>>(ei2, et2, cnt, Y, agg, r);
  }
  gemm_bf16_kernel<<<gemm_grid, TPB, 0, stream>>>(Ab, Wb + (size_t)R_N * WSTRIDE, Y); // x @ root2
  combine_kernel<<<(N_G2 * D_N + TPB - 1) / TPB, TPB, 0, stream>>>(Y, agg, xg2b, 0);

  // ---- scores + softmax ----
  score_softmax_kernel<<<L_N, S_N, 0, stream>>>(idxl, sidx, xg1, xg2b, wts, out);
}
